// PointNetSetAbstraction_12713103196705
// MI455X (gfx1250) — matrix operand order repaired and checked by reading, not compiled, not run on hardware
//
#include <hip/hip_runtime.h>
#include <hip/hip_bf16.h>
#include <stdint.h>

// ---------------------------------------------------------------------------
// PointNet++ SetAbstraction for MI455X (gfx1250, wave32).
//
// Roofline reasoning: GEMM work is only ~8.7 GFLOP but intermediates +
// output are ~350 MB -> memory bound at 23.3 TB/s. So: f16 activations /
// weights feeding v_wmma_f32_16x16x32_f16 (f32 accum), BN statistics fused
// into the GEMM epilogue (LDS ds_add_f32 + 1 global atomic per channel per
// block) so preactivations are touched exactly twice.
// ---------------------------------------------------------------------------

typedef __attribute__((ext_vector_type(16))) _Float16 v16h;
typedef __attribute__((ext_vector_type(8)))  _Float16 v8h;
typedef __attribute__((ext_vector_type(8)))  float    v8f;

#define NPTS     4096
#define NBATCH   8
#define NCENT    1024
#define KSAMP    32
#define MPOS     (NBATCH * NCENT * KSAMP)   // 262144 positions
#define RAD2     (0.2f * 0.2f)
#define BN_EPS   1e-5f
#define BN_CNT   262144.0f                  // B*K*S elements per channel

// ---------------------------------------------------------------------------
// 1) Farthest point sampling: one block per batch, distances live in LDS.
//    Writes new_xyz [B,3,S] directly into d_out.
// ---------------------------------------------------------------------------
__global__ void __launch_bounds__(256)
fps_kernel(const float* __restrict__ xyz, float* __restrict__ newxyz) {
  const int b = blockIdx.x, tid = threadIdx.x;
  __shared__ float dist[NPTS];
  __shared__ float rv[256];
  __shared__ int   ri[256];
  __shared__ int   cur;
  const float* px = xyz + (size_t)b * 3 * NPTS;
  for (int n = tid; n < NPTS; n += 256) dist[n] = 1e10f;
  if (tid == 0) cur = 0;
  __syncthreads();
  for (int it = 0; it < NCENT; ++it) {
    const int far = cur;
    if (tid == 0) {
      newxyz[(b * 3 + 0) * NCENT + it] = px[far];
      newxyz[(b * 3 + 1) * NCENT + it] = px[NPTS + far];
      newxyz[(b * 3 + 2) * NCENT + it] = px[2 * NPTS + far];
    }
    const float cx = px[far], cy = px[NPTS + far], cz = px[2 * NPTS + far];
    float best = -1.0f; int bi = 0;
    for (int n = tid; n < NPTS; n += 256) {
      const float dx = px[n] - cx, dy = px[NPTS + n] - cy, dz = px[2 * NPTS + n] - cz;
      const float d = dx * dx + dy * dy + dz * dz;
      const float dd = fminf(dist[n], d);
      dist[n] = dd;
      if (dd > best) { best = dd; bi = n; }
    }
    rv[tid] = best; ri[tid] = bi;
    __syncthreads();
    for (int s2 = 128; s2 > 0; s2 >>= 1) {
      if (tid < s2 && rv[tid + s2] > rv[tid]) { rv[tid] = rv[tid + s2]; ri[tid] = ri[tid + s2]; }
      __syncthreads();
    }
    if (tid == 0) cur = ri[0];
    __syncthreads();
  }
}

// ---------------------------------------------------------------------------
// 2) Ball query: one wave32 per centroid, ballot+prefix-popcount appends the
//    first 32 in-radius indices in index order (matches reference sort).
// ---------------------------------------------------------------------------
__global__ void __launch_bounds__(128)
ball_query_kernel(const float* __restrict__ xyz, const float* __restrict__ newxyz,
                  int* __restrict__ ball_idx) {
  const int wave = threadIdx.x >> 5, lane = threadIdx.x & 31;
  const int cid = blockIdx.x * 4 + wave;          // 0 .. B*S-1
  const int b = cid >> 10, s = cid & 1023;
  const float* px = xyz + (size_t)b * 3 * NPTS;
  const float qx = newxyz[(b * 3 + 0) * NCENT + s];
  const float qy = newxyz[(b * 3 + 1) * NCENT + s];
  const float qz = newxyz[(b * 3 + 2) * NCENT + s];
  __shared__ int slots[4][KSAMP];
  int cnt = 0;
  for (int n0 = 0; n0 < NPTS && cnt < KSAMP; n0 += 32) {
    const int n = n0 + lane;
    const float dx = px[n] - qx, dy = px[NPTS + n] - qy, dz = px[2 * NPTS + n] - qz;
    const bool in = (dx * dx + dy * dy + dz * dz) < RAD2;
    const unsigned mask = (unsigned)__ballot(in);
    if (in) {
      const int rank = cnt + __popc(mask & ((1u << lane) - 1u));
      if (rank < KSAMP) slots[wave][rank] = n;
    }
    cnt += __popc(mask);
  }
  if (cnt > KSAMP) cnt = KSAMP;
  __builtin_amdgcn_wave_barrier();
  int v;
  if (cnt == 0) v = 0;
  else          v = (lane < cnt) ? slots[wave][lane] : slots[wave][0];
  ball_idx[(size_t)cid * KSAMP + lane] = v;
}

// ---------------------------------------------------------------------------
// 3) Gather + concat -> f16 feature matrix [M, 96] (channels 67..95 zero-pad
//    so layer-0 K is a multiple of 32 for WMMA).
// ---------------------------------------------------------------------------
__global__ void __launch_bounds__(256)
gather_kernel(const float* __restrict__ xyz, const float* __restrict__ points,
              const float* __restrict__ newxyz, const int* __restrict__ ball_idx,
              _Float16* __restrict__ feat) {
  const int gid = blockIdx.x * 256 + threadIdx.x;      // M*96 threads
  const int p = gid / 96;
  const int c = gid - p * 96;
  const int b = p >> 15;
  const int rem = p & 32767;
  const int s = rem >> 5;
  const int n = ball_idx[p];
  float v;
  if (c < 3)       v = xyz[((size_t)b * 3 + c) * NPTS + n] - newxyz[(b * 3 + c) * NCENT + s];
  else if (c < 67) v = points[((size_t)b * 64 + (c - 3)) * NPTS + n];
  else             v = 0.0f;
  feat[(size_t)p * 96 + c] = (_Float16)v;
}

// ---------------------------------------------------------------------------
// 4) f16 GEMM with WMMA + fused bias + fused BN statistics.
//    C[M,N] = A[M,LDK] * W[N,LDK]^T + bias.  One wave computes 16 rows x N.
//    KT = K/32 k-steps, NT = N/16 column tiles, all compile-time.
// ---------------------------------------------------------------------------
template <int KT, int NT, int LDK>
__global__ void __launch_bounds__(256)
gemm_bn_kernel(const _Float16* __restrict__ A, const _Float16* __restrict__ W,
               const float* __restrict__ bias, float* __restrict__ C,
               float* __restrict__ gsums /* [0..127]=sum, [128..255]=sumsq */) {
  constexpr int N = NT * 16;
  __shared__ float ssum[N];
  __shared__ float ssq[N];
  const int tid = threadIdx.x;
  if (tid < N) { ssum[tid] = 0.0f; ssq[tid] = 0.0f; }
  __syncthreads();

  const int wave = tid >> 5, lane = tid & 31;
  const int lr = lane & 15;           // row within A tile / column within B tile
  const int hi = lane >> 4;           // half-wave select per ISA fragment layout
  const int row0 = blockIdx.x * 128 + wave * 16;

  v8f acc[NT];
  const v8f vzero = {0.f, 0.f, 0.f, 0.f, 0.f, 0.f, 0.f, 0.f};
#pragma unroll
  for (int nt = 0; nt < NT; ++nt) acc[nt] = vzero;

#pragma unroll
  for (int kt = 0; kt < KT; ++kt) {
    // A fragment (16x32 f16): lane<16 holds K {0..15},
    // lane>=16 holds K {16..31} of row (lane&15).
    const _Float16* arow = A + (size_t)(row0 + lr) * LDK + kt * 32 + hi * 16;
    v16h afrag;
    ((v8h*)&afrag)[0] = *(const v8h*)(arow);
    ((v8h*)&afrag)[1] = *(const v8h*)(arow + 8);
#pragma unroll
    for (int nt = 0; nt < NT; ++nt) {
      // B fragment (32x16 f16): column o = nt*16+lr holds 16 contiguous
      // K-values starting at hi*16 — exactly row-major W[o][k].
      const _Float16* wrow = W + (size_t)(nt * 16 + lr) * LDK + kt * 32 + hi * 16;
      const v16h bfrag = *(const v16h*)(wrow);
      acc[nt] = __builtin_amdgcn_wmma_f32_16x16x32_f16(
          false, afrag, false, bfrag, (short)0, acc[nt], false, false);
    }
  }

  // Epilogue: bias, BN partial sums (LDS ds_add_f32), store preactivations.
#pragma unroll
  for (int nt = 0; nt < NT; ++nt) {
    const int o = nt * 16 + lr;
    const float bv = bias[o];
    float ls = 0.0f, lq = 0.0f;
#pragma unroll
    for (int i = 0; i < 8; ++i) {
      const float v = acc[nt][i] + bv;
      acc[nt][i] = v;
      ls += v; lq += v * v;
    }
    atomicAdd(&ssum[o], ls);
    atomicAdd(&ssq[o], lq);
#pragma unroll
    for (int i = 0; i < 8; ++i) {
      const int m = row0 + i + hi * 8;          // C/D layout: VGPR i -> row i or 8+i
      C[(size_t)m * N + o] = acc[nt][i];
    }
  }
  __syncthreads();
  if (tid < N) {
    atomicAdd(&gsums[tid], ssum[tid]);
    atomicAdd(&gsums[128 + tid], ssq[tid]);
  }
}

// ---------------------------------------------------------------------------
// 5) BN (training-mode batch stats) + ReLU.
// ---------------------------------------------------------------------------
__global__ void __launch_bounds__(256)
bn_relu_mid_kernel(const float* __restrict__ C, const float* __restrict__ gsums,
                   const float* __restrict__ g, const float* __restrict__ beta,
                   _Float16* __restrict__ act) {               // N = 64
  const int idx = blockIdx.x * 256 + threadIdx.x;              // M*64 threads
  const int o = idx & 63;
  const float mean = gsums[o] / BN_CNT;
  const float var = gsums[128 + o] / BN_CNT - mean * mean;
  const float inv = rsqrtf(var + BN_EPS);
  const float x = C[(size_t)idx];
  float y = g[o] * (x - mean) * inv + beta[o];
  y = y > 0.0f ? y : 0.0f;
  act[(size_t)idx] = (_Float16)y;
}

__global__ void __launch_bounds__(256)
bn_relu_final_kernel(const float* __restrict__ C, const float* __restrict__ gsums,
                     const float* __restrict__ g, const float* __restrict__ beta,
                     float* __restrict__ out) {                // N = 128, [B,128,K,S]
  const int idx = blockIdx.x * 256 + threadIdx.x;              // B*128*K*S threads
  const int s = idx & 1023;
  const int k = (idx >> 10) & 31;
  const int o = (idx >> 15) & 127;
  const int b = idx >> 22;
  const int p = ((b << 10) + s) * KSAMP + k;                   // position index
  const float mean = gsums[o] / BN_CNT;
  const float var = gsums[128 + o] / BN_CNT - mean * mean;
  const float inv = rsqrtf(var + BN_EPS);
  const float x = C[(size_t)p * 128 + o];
  float y = g[o] * (x - mean) * inv + beta[o];
  out[(size_t)idx] = y > 0.0f ? y : 0.0f;                      // coalesced store
}

// ---------------------------------------------------------------------------
// Helpers: f32 -> f16 weight conversion (with K padding) and stats zeroing.
// ---------------------------------------------------------------------------
__global__ void __launch_bounds__(256)
convert_w_kernel(const float* __restrict__ w, _Float16* __restrict__ wh,
                 int cout, int cin, int ld) {
  const int i = blockIdx.x * 256 + threadIdx.x;
  if (i >= cout * ld) return;
  const int o = i / ld, c = i - o * ld;
  wh[i] = (_Float16)(c < cin ? w[o * cin + c] : 0.0f);
}

__global__ void zero256_kernel(float* __restrict__ p) { p[threadIdx.x] = 0.0f; }

// ---------------------------------------------------------------------------
// Host launcher.
// ---------------------------------------------------------------------------
extern "C" void kernel_launch(void* const* d_in, const int* in_sizes, int n_in,
                              void* d_out, int out_size, void* d_ws, size_t ws_size,
                              hipStream_t stream) {
  const float* xyz   = (const float*)d_in[0];
  const float* pts   = (const float*)d_in[1];
  const float* w0 = (const float*)d_in[2],  *b0 = (const float*)d_in[3];
  const float* g0 = (const float*)d_in[4],  *be0 = (const float*)d_in[5];
  const float* w1 = (const float*)d_in[6],  *b1 = (const float*)d_in[7];
  const float* g1 = (const float*)d_in[8],  *be1 = (const float*)d_in[9];
  const float* w2 = (const float*)d_in[10], *b2 = (const float*)d_in[11];
  const float* g2 = (const float*)d_in[12], *be2 = (const float*)d_in[13];
  float* out = (float*)d_out;                    // [B,3,S] ++ [B,128,K,S]

  char* ws = (char*)d_ws;
  int*      ball_idx = (int*)(ws + 0);                        //   1,048,576 B
  _Float16* featA    = (_Float16*)(ws + 1048576ull);          //  50,331,648 B  (M x 96 f16)
  _Float16* actB     = (_Float16*)(ws + 51380224ull);         //  33,554,432 B  (M x 64 f16)
  float*    preact   = (float*)(ws + 84934656ull);            // 134,217,728 B  (M x 128 f32)
  _Float16* w0h      = (_Float16*)(ws + 219152384ull);        //      12,288 B
  _Float16* w1h      = (_Float16*)(ws + 219164672ull);        //       8,192 B
  _Float16* w2h      = (_Float16*)(ws + 219172864ull);        //      16,384 B
  float*    gsums    = (float*)(ws + 219189248ull);           //       1,024 B
  if (ws_size < 219190272ull) return;

  // Stage 1: FPS (writes new_xyz part of d_out) and ball query.
  fps_kernel<<<NBATCH, 256, 0, stream>>>(xyz, out);
  ball_query_kernel<<<(NBATCH * NCENT) / 4, 128, 0, stream>>>(xyz, out, ball_idx);

  // Weight conversion (tiny; overlaps nothing it depends on).
  convert_w_kernel<<<(64 * 96 + 255) / 256, 256, 0, stream>>>(w0, w0h, 64, 67, 96);
  convert_w_kernel<<<(64 * 64 + 255) / 256, 256, 0, stream>>>(w1, w1h, 64, 64, 64);
  convert_w_kernel<<<(128 * 64 + 255) / 256, 256, 0, stream>>>(w2, w2h, 128, 64, 64);

  // Stage 2: gather/concat into padded f16 features.
  gather_kernel<<<(MPOS * 96) / 256, 256, 0, stream>>>(xyz, pts, out, ball_idx, featA);

  // Layer 0: 96 -> 64
  zero256_kernel<<<1, 256, 0, stream>>>(gsums);
  gemm_bn_kernel<3, 4, 96><<<MPOS / 128, 256, 0, stream>>>(featA, w0h, b0, preact, gsums);
  bn_relu_mid_kernel<<<(MPOS * 64) / 256, 256, 0, stream>>>(preact, gsums, g0, be0, actB);

  // Layer 1: 64 -> 64
  zero256_kernel<<<1, 256, 0, stream>>>(gsums);
  gemm_bn_kernel<2, 4, 64><<<MPOS / 128, 256, 0, stream>>>(actB, w1h, b1, preact, gsums);
  bn_relu_mid_kernel<<<(MPOS * 64) / 256, 256, 0, stream>>>(preact, gsums, g1, be1, actB);

  // Layer 2: 64 -> 128, fused transpose to [B,128,K,S] in d_out.
  zero256_kernel<<<1, 256, 0, stream>>>(gsums);
  gemm_bn_kernel<2, 8, 64><<<MPOS / 128, 256, 0, stream>>>(actB, w2h, b2, preact, gsums);
  bn_relu_final_kernel<<<(MPOS * 128) / 256, 256, 0, stream>>>(preact, gsums, g2, be2,
                                                               out + NBATCH * 3 * NCENT);
}
